// TopKAutoEncoder_9113920602204
// MI455X (gfx1250) — matrix operand order repaired and checked by reading, not compiled, not run on hardware
//
#include <hip/hip_runtime.h>
#include <hip/hip_bf16.h>

// ---------------------------------------------------------------------------
// TopK sparse autoencoder forward, MI455X (gfx1250, wave32, WMMA).
// Encoder GEMM in split-bf16 (hi/lo) via V_WMMA_F32_16X16X32_BF16 with fused
// per-row running top-32 selection; sparse decode + FVU.
// ---------------------------------------------------------------------------

#define B_ROWS 8192
#define D_IN   1024
#define N_DICT 32768
#define K_TOP  32

#define BM      64                 // rows per block
#define BN      128                // cols per inner tile
#define NCHUNK  4096               // cols per block (outer loop)
#define NITER   (NCHUNK / BN)      // 32
#define NCHUNKS (N_DICT / NCHUNK)  // 8
#define KSTEP   32
#define TPAD    132
#define NEG_HUGE (-3.402823466e38f)

typedef __bf16 bf16_t;
typedef __bf16 v16bf __attribute__((ext_vector_type(16)));
typedef __bf16 v8bf  __attribute__((ext_vector_type(8)));
typedef float  v8f   __attribute__((ext_vector_type(8)));

// ---------------------------------------------------------------------------
// prep_x: xc = x - b_dec -> bf16 hi/lo split; per-block colsum / sum(x^2)
// grid: 512 blocks x 256 threads (16 rows/block, 4 cols/thread)
// ---------------------------------------------------------------------------
__launch_bounds__(256)
__global__ void sae_prep_x_kernel(const float* __restrict__ x,
                                  const float* __restrict__ b_dec,
                                  bf16_t* __restrict__ XH, bf16_t* __restrict__ XL,
                                  float* __restrict__ colpart,
                                  float* __restrict__ sspart)
{
    __shared__ float red[8];
    const int tid  = threadIdx.x;
    const int lane = tid & 31, wave = tid >> 5;
    const int c0   = tid * 4;
    const int row0 = blockIdx.x * 16;

    float4 bd = *(const float4*)(b_dec + c0);
    float cs[4] = {0.f, 0.f, 0.f, 0.f};
    float bb[4] = {bd.x, bd.y, bd.z, bd.w};
    float ss = 0.f;

    for (int r = 0; r < 16; ++r) {
        size_t off = (size_t)(row0 + r) * D_IN + c0;
        float4 xv = *(const float4*)(x + off);
        float v[4] = {xv.x, xv.y, xv.z, xv.w};
        #pragma unroll
        for (int i = 0; i < 4; ++i) {
            cs[i] += v[i];
            ss    += v[i] * v[i];
            float c = v[i] - bb[i];
            bf16_t h = (bf16_t)c;
            XH[off + i] = h;
            XL[off + i] = (bf16_t)(c - (float)h);
        }
    }
    #pragma unroll
    for (int i = 0; i < 4; ++i)
        colpart[(size_t)blockIdx.x * D_IN + c0 + i] = cs[i];

    #pragma unroll
    for (int off = 16; off >= 1; off >>= 1) ss += __shfl_xor(ss, off, 32);
    if (lane == 0) red[wave] = ss;
    __syncthreads();
    if (tid == 0) {
        float s = 0.f;
        for (int i = 0; i < 8; ++i) s += red[i];
        sspart[blockIdx.x] = s;
    }
}

// ---------------------------------------------------------------------------
// prep_w: split W_enc (fp32) -> bf16 hi/lo.  grid: 32768 x 256, 4 elem/thread
// ---------------------------------------------------------------------------
__launch_bounds__(256)
__global__ void sae_prep_w_kernel(const float* __restrict__ W,
                                  bf16_t* __restrict__ WH, bf16_t* __restrict__ WL)
{
    size_t i = ((size_t)blockIdx.x * 256 + threadIdx.x) * 4;
    float4 w = *(const float4*)(W + i);
    float v[4] = {w.x, w.y, w.z, w.w};
    #pragma unroll
    for (int j = 0; j < 4; ++j) {
        bf16_t h = (bf16_t)v[j];
        WH[i + j] = h;
        WL[i + j] = (bf16_t)(v[j] - (float)h);
    }
}

// ---------------------------------------------------------------------------
// gemm_topk: pre_acts = relu(xc @ W_enc^T + b_enc) tile-by-tile via WMMA
// (bf16x3 split), fused per-row running top-32 over a 4096-column chunk.
// grid: (8192/64, 32768/4096) = (128, 8), 256 threads (8 waves: 4M x 2N).
// ---------------------------------------------------------------------------
__launch_bounds__(256)
__global__ void sae_gemm_topk_kernel(const bf16_t* __restrict__ XH,
                                     const bf16_t* __restrict__ XL,
                                     const bf16_t* __restrict__ WH,
                                     const bf16_t* __restrict__ WL,
                                     const float* __restrict__ b_enc,
                                     float* __restrict__ cand_val,
                                     int*   __restrict__ cand_idx)
{
    __shared__ float tile[BM][TPAD];
    __shared__ float list_val[BM][32];
    __shared__ int   list_idx[BM][32];
    __shared__ float thresh[BM];

    const int tid  = threadIdx.x;
    const int lane = tid & 31;
    const int wave = tid >> 5;
    const int mw   = wave >> 1;  // 0..3
    const int nw   = wave & 1;   // 0..1
    const int bm   = blockIdx.x * BM;
    const int nchunk_base = blockIdx.y * NCHUNK;

    // init top-k lists
    for (int r = wave; r < BM; r += 8) {
        list_val[r][lane] = NEG_HUGE;
        list_idx[r][lane] = 0;
        if (lane == 0) thresh[r] = NEG_HUGE;
    }
    __syncthreads();

    // A-fragment addressing (16x32 tile, contiguous k: lane half h holds k = 16h + s, same as B)
    const int arow = bm + mw * 16 + (lane & 15);
    const int asel = (lane >> 4) * 16;  // K sub-chunk 0 or 16
    const bf16_t* aph = XH + (size_t)arow * D_IN + asel;
    const bf16_t* apl = XL + (size_t)arow * D_IN + asel;

    const int bcol = lane & 15;
    const int kc   = (lane >> 4) * 16;  // B K chunk 0 or 16
    const int crow_half = (lane >> 4) * 8;
    const int ccol = lane & 15;

    for (int nt = 0; nt < NITER; ++nt) {
        const int nbase = nchunk_base + nt * BN + nw * 64;  // wave's first col

        const bf16_t* bph[4];
        const bf16_t* bpl[4];
        #pragma unroll
        for (int f = 0; f < 4; ++f) {
            size_t off = (size_t)(nbase + f * 16 + bcol) * D_IN + kc;
            bph[f] = WH + off;
            bpl[f] = WL + off;
        }

        v8f acc[4];
        #pragma unroll
        for (int f = 0; f < 4; ++f)
            #pragma unroll
            for (int r = 0; r < 8; ++r) acc[f][r] = 0.f;

        #pragma unroll 2
        for (int ks = 0; ks < D_IN; ks += KSTEP) {
            v8bf a0 = *(const v8bf*)(aph + ks);
            v8bf a1 = *(const v8bf*)(aph + ks + 8);
            v16bf ah = __builtin_shufflevector(a0, a1, 0,1,2,3,4,5,6,7,8,9,10,11,12,13,14,15);
            v8bf a2 = *(const v8bf*)(apl + ks);
            v8bf a3 = *(const v8bf*)(apl + ks + 8);
            v16bf al = __builtin_shufflevector(a2, a3, 0,1,2,3,4,5,6,7,8,9,10,11,12,13,14,15);

            v16bf bh[4], bl[4];
            #pragma unroll
            for (int f = 0; f < 4; ++f) {
                bh[f] = *(const v16bf*)(bph[f] + ks);
                bl[f] = *(const v16bf*)(bpl[f] + ks);
            }
            if (ks + KSTEP < D_IN) {
                __builtin_prefetch((const void*)(bph[0] + ks + KSTEP), 0, 3);
                __builtin_prefetch((const void*)(aph + ks + KSTEP), 0, 3);
            }
            #pragma unroll
            for (int f = 0; f < 4; ++f)
                acc[f] = __builtin_amdgcn_wmma_f32_16x16x32_bf16(
                    false, ah, false, bh[f], (short)0, acc[f], false, false);
            #pragma unroll
            for (int f = 0; f < 4; ++f)
                acc[f] = __builtin_amdgcn_wmma_f32_16x16x32_bf16(
                    false, ah, false, bl[f], (short)0, acc[f], false, false);
            #pragma unroll
            for (int f = 0; f < 4; ++f)
                acc[f] = __builtin_amdgcn_wmma_f32_16x16x32_bf16(
                    false, al, false, bh[f], (short)0, acc[f], false, false);
        }

        // bias + relu, dump tile to LDS (C layout: VGPR r -> M=r(+8), lane%16 -> N)
        #pragma unroll
        for (int f = 0; f < 4; ++f) {
            int coll = nw * 64 + f * 16 + ccol;  // 0..127
            float bias = b_enc[nchunk_base + nt * BN + coll];
            #pragma unroll
            for (int r = 0; r < 8; ++r) {
                float v = acc[f][r] + bias;
                v = v > 0.f ? v : 0.f;
                tile[mw * 16 + crow_half + r][coll] = v;
            }
        }
        __syncthreads();

        // running top-32: each wave owns 8 rows, threshold-filtered insertion
        const int gcol_base = nchunk_base + nt * BN;
        #pragma unroll 1
        for (int i = 0; i < 8; ++i) {
            const int r = wave * 8 + i;
            float t = thresh[r];
            #pragma unroll 1
            for (int j = 0; j < 4; ++j) {
                float v = tile[r][lane + 32 * j];
                unsigned mask = __builtin_amdgcn_ballot_w32(v > t);
                while (mask) {
                    int src = __ffs(mask) - 1;
                    mask &= mask - 1;
                    float vv = __shfl(v, src, 32);
                    int gidx = gcol_base + src + 32 * j;
                    // argmin over the 32-entry list (one entry per lane)
                    float lv = list_val[r][lane];
                    float m = lv; int ml = lane;
                    #pragma unroll
                    for (int off = 16; off >= 1; off >>= 1) {
                        float om = __shfl_xor(m, off, 32);
                        int  oml = __shfl_xor(ml, off, 32);
                        if (om < m || (om == m && oml < ml)) { m = om; ml = oml; }
                    }
                    if (vv > m) {
                        if (lane == ml) { list_val[r][lane] = vv; list_idx[r][lane] = gidx; }
                        float lv2 = (lane == ml) ? vv : lv;
                        float m2 = lv2;
                        #pragma unroll
                        for (int off = 16; off >= 1; off >>= 1) {
                            float om = __shfl_xor(m2, off, 32);
                            m2 = m2 < om ? m2 : om;
                        }
                        t = m2;  // tightened threshold
                    }
                }
            }
            if (lane == 0) thresh[r] = t;
        }
        __syncthreads();
    }

    // emit 32 candidates per (row, chunk)
    #pragma unroll 1
    for (int i = 0; i < 8; ++i) {
        const int r = wave * 8 + i;
        const size_t base = (size_t)(bm + r) * (NCHUNKS * 32) + (size_t)blockIdx.y * 32;
        cand_val[base + lane] = list_val[r][lane];
        cand_idx[base + lane] = list_idx[r][lane];
    }
}

// ---------------------------------------------------------------------------
// merge_topk: per row, top-32 of 256 candidates (sorted desc, like lax.top_k).
// grid: 1024 blocks x 256 threads (one wave per row)
// ---------------------------------------------------------------------------
__launch_bounds__(256)
__global__ void sae_merge_topk_kernel(const float* __restrict__ cand_val,
                                      const int*   __restrict__ cand_idx,
                                      float* __restrict__ top_acts,
                                      int*   __restrict__ top_idx)
{
    const int lane = threadIdx.x & 31;
    const int wave = threadIdx.x >> 5;
    const int row  = blockIdx.x * 8 + wave;
    const float* cv = cand_val + (size_t)row * (NCHUNKS * 32);
    const int*   ci = cand_idx + (size_t)row * (NCHUNKS * 32);

    float v[8]; int id[8];
    #pragma unroll
    for (int s = 0; s < 8; ++s) { v[s] = cv[lane + 32 * s]; id[s] = ci[lane + 32 * s]; }

    float* oa = top_acts + (size_t)row * K_TOP;
    int*   oi = top_idx  + (size_t)row * K_TOP;

    for (int k = 0; k < K_TOP; ++k) {
        float bmv = v[0]; int bs = 0;
        #pragma unroll
        for (int s = 1; s < 8; ++s)
            if (v[s] > bmv) { bmv = v[s]; bs = s; }
        int bpos = lane + 32 * bs;
        #pragma unroll
        for (int off = 16; off >= 1; off >>= 1) {
            float om = __shfl_xor(bmv, off, 32);
            int   op = __shfl_xor(bpos, off, 32);
            if (om > bmv || (om == bmv && op < bpos)) { bmv = om; bpos = op; }
        }
        const int owner = bpos & 31, slot = bpos >> 5;
        if (lane == owner) {
            oa[k] = bmv;
            oi[k] = id[slot];
            v[slot] = NEG_HUGE;
        }
    }
}

// ---------------------------------------------------------------------------
// decode: sae_out = sum_k act*W_dec[idx] + b_dec; per-row error^2 partial.
// grid: 8192 blocks x 256 threads (4 cols/thread)
// ---------------------------------------------------------------------------
__launch_bounds__(256)
__global__ void sae_decode_kernel(const float* __restrict__ x,
                                  const float* __restrict__ W_dec,
                                  const float* __restrict__ b_dec,
                                  const float* __restrict__ top_acts,
                                  const int*   __restrict__ top_idx,
                                  float* __restrict__ sae_out,
                                  float* __restrict__ ssepart)
{
    __shared__ float sacts[K_TOP];
    __shared__ int   sidx[K_TOP];
    __shared__ float red[8];
    const int row = blockIdx.x;
    const int tid = threadIdx.x;
    const int lane = tid & 31, wave = tid >> 5;
    if (tid < K_TOP) {
        sacts[tid] = top_acts[(size_t)row * K_TOP + tid];
        sidx[tid]  = top_idx[(size_t)row * K_TOP + tid];
    }
    __syncthreads();

    const int c0 = tid * 4;
    float4 acc = *(const float4*)(b_dec + c0);
    #pragma unroll 4
    for (int j = 0; j < K_TOP; ++j) {
        const float a = sacts[j];
        const float4 w = *(const float4*)(W_dec + (size_t)sidx[j] * D_IN + c0);
        acc.x += a * w.x; acc.y += a * w.y; acc.z += a * w.z; acc.w += a * w.w;
    }
    *(float4*)(sae_out + (size_t)row * D_IN + c0) = acc;

    float4 xv = *(const float4*)(x + (size_t)row * D_IN + c0);
    float ex = acc.x - xv.x, ey = acc.y - xv.y, ez = acc.z - xv.z, ew = acc.w - xv.w;
    float ss = ex * ex + ey * ey + ez * ez + ew * ew;
    #pragma unroll
    for (int off = 16; off >= 1; off >>= 1) ss += __shfl_xor(ss, off, 32);
    if (lane == 0) red[wave] = ss;
    __syncthreads();
    if (tid == 0) {
        float s = 0.f;
        for (int i = 0; i < 8; ++i) s += red[i];
        ssepart[row] = s;
    }
}

// ---------------------------------------------------------------------------
// finalize: total_variance = sum x^2 - (1/B)*sum_d colsum_d^2 ; fvu = sse/tv
// ---------------------------------------------------------------------------
__launch_bounds__(256)
__global__ void sae_finalize_kernel(const float* __restrict__ colpart,
                                    const float* __restrict__ sspart,
                                    const float* __restrict__ ssepart,
                                    float* __restrict__ out_scalars)
{
    __shared__ float red[8][3];
    const int tid = threadIdx.x;
    const int lane = tid & 31, wave = tid >> 5;

    float csq = 0.f;
    for (int d = tid; d < D_IN; d += 256) {
        float s = 0.f;
        for (int b = 0; b < B_ROWS / 16; ++b) s += colpart[(size_t)b * D_IN + d];
        csq += s * s;
    }
    float ssx = 0.f;
    for (int i = tid; i < B_ROWS / 16; i += 256) ssx += sspart[i];
    float sse = 0.f;
    for (int i = tid; i < B_ROWS; i += 256) sse += ssepart[i];

    #pragma unroll
    for (int off = 16; off >= 1; off >>= 1) {
        csq += __shfl_xor(csq, off, 32);
        ssx += __shfl_xor(ssx, off, 32);
        sse += __shfl_xor(sse, off, 32);
    }
    if (lane == 0) { red[wave][0] = csq; red[wave][1] = ssx; red[wave][2] = sse; }
    __syncthreads();
    if (tid == 0) {
        float a = 0.f, b = 0.f, c = 0.f;
        for (int i = 0; i < 8; ++i) { a += red[i][0]; b += red[i][1]; c += red[i][2]; }
        float tv = b - a * (1.0f / (float)B_ROWS);
        if (tv == 0.0f) tv = 1.0f;
        out_scalars[0] = c / tv;  // fvu
        out_scalars[1] = 0.0f;    // auxk_loss
        out_scalars[2] = 0.0f;    // multi_topk_fvu
    }
}

// ---------------------------------------------------------------------------
extern "C" void kernel_launch(void* const* d_in, const int* in_sizes, int n_in,
                              void* d_out, int out_size, void* d_ws, size_t ws_size,
                              hipStream_t stream)
{
    const float* x     = (const float*)d_in[0];
    const float* W_enc = (const float*)d_in[1];
    const float* b_enc = (const float*)d_in[2];
    const float* W_dec = (const float*)d_in[3];
    const float* b_dec = (const float*)d_in[4];
    // d_in[5] = k (compile-time K_TOP=32)

    char* ws = (char*)d_ws;
    bf16_t* XH = (bf16_t*)(ws + 0);
    bf16_t* XL = (bf16_t*)(ws + 16777216);
    bf16_t* WH = (bf16_t*)(ws + 33554432);
    bf16_t* WL = (bf16_t*)(ws + 100663296);
    float* cand_val = (float*)(ws + 167772160);
    int*   cand_idx = (int*)  (ws + 176160768);
    float* colpart  = (float*)(ws + 184549376);
    float* sspart   = (float*)(ws + 186646528);
    float* ssepart  = (float*)(ws + 186648576);

    float* sae_out  = (float*)d_out;
    float* top_acts = sae_out + (size_t)B_ROWS * D_IN;
    int*   top_idx  = (int*)(top_acts + (size_t)B_ROWS * K_TOP);
    float* scalars  = (float*)(top_idx + (size_t)B_ROWS * K_TOP);

    sae_prep_x_kernel<<<B_ROWS / 16, 256, 0, stream>>>(x, b_dec, XH, XL, colpart, sspart);
    sae_prep_w_kernel<<<(N_DICT * (D_IN / 4)) / 256, 256, 0, stream>>>(W_enc, WH, WL);

    dim3 ggrid(B_ROWS / BM, N_DICT / NCHUNK);  // 128 x 8
    sae_gemm_topk_kernel<<<ggrid, 256, 0, stream>>>(XH, XL, WH, WL, b_enc, cand_val, cand_idx);

    sae_merge_topk_kernel<<<B_ROWS / 8, 256, 0, stream>>>(cand_val, cand_idx, top_acts, top_idx);
    sae_decode_kernel<<<B_ROWS, 256, 0, stream>>>(x, W_dec, b_dec, top_acts, top_idx, sae_out, ssepart);
    sae_finalize_kernel<<<1, 256, 0, stream>>>(colpart, sspart, ssepart, scalars);
}
